// RNN_429496730073
// MI455X (gfx1250) — matrix operand order repaired and checked by reading, not compiled, not run on hardware
//
#include <hip/hip_runtime.h>

// ---------------------------------------------------------------------------
// Fused LSTM (B=4096, T=256, I=30, H=128) + MLP head for MI455X (gfx1250).
//
// Final form (= v3, empirically the cleanest codegen):
//  - One persistent workgroup per 32 batch rows (128 blocks x 16 wave32).
//  - W_hh/W_ih staged once into LDS, then all 20 WMMA B-fragments hoisted
//    into VGPRs for the whole 256-step loop; the large static LDS also pins
//    the backend to 1-block occupancy, which grants the full VGPR budget
//    (small-LDS variants made the allocator spill the fragments and stream
//    them from scratch every step).
//  - Bias folded into augmented W_ih cols 30/31 (two-term bf16 split) with
//    x cols 30/31 = 1.0 -> accumulators start from the inline-0 C operand.
//  - Hardware v_tanh_f32 activations; h double-buffered in LDS, one barrier
//    per timestep; c-state lives in VGPRs.
// ---------------------------------------------------------------------------

#define TSEQ    256
#define IN      30
#define INP     32      // K padded: col 30 = bias_hi, col 31 = bias_lo
#define INP_PAD 40      // LDS row stride (bank rotation)
#define H       128
#define H_PAD   136     // LDS row stride (bank rotation)
#define G4      512     // 4*H gates
#define NOUT    11
#define MTILE   32      // batch rows per workgroup (two M=16 tiles)
#define THREADS 512     // 16 waves (wave32)

typedef __attribute__((ext_vector_type(16))) __bf16 v16bf;
typedef __attribute__((ext_vector_type(8)))  __bf16 v8bf;
typedef __attribute__((ext_vector_type(8)))  float  v8f;

static constexpr size_t SMEM_BYTES =
    (size_t)G4 * H_PAD * 2 +            // Whh   [512][H_PAD] bf16
    (size_t)G4 * INP_PAD * 2 +          // Wih   [512][INP_PAD] bf16 (reused: W0)
    G4 * 4 +                            // bias  [512] f32 (head: b0)
    2 * (size_t)MTILE * H_PAD * 2 +     // H     [2][32][H_PAD] bf16 (double buf)
    2 * (size_t)MTILE * INP_PAD * 2 +   // X     [2][32][INP_PAD] bf16
    (size_t)MTILE * H_PAD * 2 +         // Y0    [32][H_PAD] bf16
    16 * H_PAD * 2 +                    // W1    [16][H_PAD] bf16
    16 * 4;                             // b1    [16] f32

__device__ __forceinline__ float fast_tanh(float x) {
#if __has_builtin(__builtin_amdgcn_tanhf)
  return __builtin_amdgcn_tanhf(x);     // v_tanh_f32 (CDNA5 TRANS op)
#else
  x = fminf(fmaxf(x, -15.0f), 15.0f);
  float e = __expf(-2.0f * x);
  return (1.0f - e) / (1.0f + e);
#endif
}
__device__ __forceinline__ float fast_sigm(float x) {
  return fmaf(0.5f, fast_tanh(0.5f * x), 0.5f);
}

// 16-bit A-matrix 16x32 fragment (ISA 7.12.2): lane (m = lane%16, hi = lane/16)
// holds K = k0 + hi*8 .. +8 and K = k0 + 16 + hi*8 .. +8 (two 16B chunks).
__device__ __forceinline__ v16bf load_a_frag(const __bf16* row, int k0, int hi) {
  v8bf lo = *(const v8bf*)(row + k0 + hi * 8);
  v8bf hh = *(const v8bf*)(row + k0 + 16 + hi * 8);
  return __builtin_shufflevector(lo, hh, 0, 1, 2, 3, 4, 5, 6, 7,
                                         8, 9, 10, 11, 12, 13, 14, 15);
}
// 16-bit B-matrix 32x16 fragment: lane (n = lane%16, hi = lane/16) holds
// K = k0 + hi*8 .. +8 and K = k0 + 16 + hi*8 .. +8 of column n. `row` = &W[n][0] (k-major).
__device__ __forceinline__ v16bf load_b_frag(const __bf16* row, int k0, int hi) {
  v8bf lo = *(const v8bf*)(row + k0 + hi * 8);
  v8bf hh = *(const v8bf*)(row + k0 + 16 + hi * 8);
  return __builtin_shufflevector(lo, hh, 0, 1, 2, 3, 4, 5, 6, 7,
                                         8, 9, 10, 11, 12, 13, 14, 15);
}

#define WMMA_BF16(A, B, C) \
  __builtin_amdgcn_wmma_f32_16x16x32_bf16(false, (A), false, (B), (short)0, (C), false, false)

__global__ __launch_bounds__(THREADS, 1)
void lstm_fused_kernel(const float* __restrict__ x,
                       const float* __restrict__ W_ih,
                       const float* __restrict__ W_hh,
                       const float* __restrict__ b_ih,
                       const float* __restrict__ b_hh,
                       const float* __restrict__ W0,
                       const float* __restrict__ b0,
                       const float* __restrict__ W1,
                       const float* __restrict__ b1,
                       float* __restrict__ out)
{
  extern __shared__ char smem[];
  __bf16* ldsWhh  = (__bf16*)smem;                          // [512][H_PAD]
  __bf16* ldsWih  = ldsWhh + G4 * H_PAD;                    // [512][INP_PAD]
  float*  ldsBias = (float*)(ldsWih + G4 * INP_PAD);        // [512] (head b0)
  __bf16* ldsH    = (__bf16*)(ldsBias + G4);                // [2][32][H_PAD]
  __bf16* ldsX    = ldsH + 2 * MTILE * H_PAD;               // [2][32][INP_PAD]
  __bf16* ldsY0   = ldsX + 2 * MTILE * INP_PAD;             // [32][H_PAD]
  __bf16* ldsW1   = ldsY0 + MTILE * H_PAD;                  // [16][H_PAD]
  float*  ldsB1   = (float*)(ldsW1 + 16 * H_PAD);           // [16]

  const int tid  = threadIdx.x;
  const int wv   = tid >> 5;
  const int lane = tid & 31;
  const int mrow = lane & 15;   // A-fragment row within M-tile
  const int hi   = lane >> 4;   // lane half
  const int mt   = wv >> 3;     // which M=16 tile of the 32-row batch block
  const int ws   = wv & 7;      // h-column slice (16 cols) this wave owns
  const int ncol = lane & 15;   // column within N-tile (C-layout: col = lane%16)
  const int gcol = ws * 16 + ncol;  // column in [0,128) of the hidden dim

  // ---- stage weights into LDS (bf16); bias folded into Wih cols 30/31 ----
  for (int idx = tid; idx < G4 * H; idx += THREADS) {
    int r = idx >> 7, c = idx & 127;
    ldsWhh[r * H_PAD + c] = (__bf16)W_hh[idx];
  }
  for (int idx = tid; idx < G4 * INP; idx += THREADS) {
    int r = idx >> 5, c = idx & 31;
    __bf16 v;
    if (c < IN) {
      v = (__bf16)W_ih[r * IN + c];
    } else {
      float b = b_ih[r] + b_hh[r];
      __bf16 bhiv = (__bf16)b;
      v = (c == IN) ? bhiv : (__bf16)(b - (float)bhiv);  // hi / residual
    }
    ldsWih[r * INP_PAD + c] = v;
  }
  for (int idx = tid; idx < 2 * MTILE * H_PAD; idx += THREADS)
    ldsH[idx] = (__bf16)0.0f;

  // x prefetch plumbing: each thread owns 2 fixed (row, col) slots
  const float* xblk = x + (size_t)blockIdx.x * MTILE * TSEQ * IN;
  const int xr0 = tid >> 5;          // rows 0..15 ; +16 for the second slot
  const int xc  = tid & 31;
  const bool xcv = (xc < IN);
  const float* xsrc0 = xblk + (size_t)xr0 * TSEQ * IN + xc;
  const float* xsrc1 = xblk + (size_t)(xr0 + 16) * TSEQ * IN + xc;
  __bf16* xdst0 = ldsX + xr0 * INP_PAD + xc;
  __bf16* xdst1 = ldsX + (xr0 + 16) * INP_PAD + xc;

  // preload x[t=0] into buffer 0 (pad cols 30/31 with 1.0 for the bias trick)
  xdst0[0] = xcv ? (__bf16)xsrc0[0] : (__bf16)1.0f;
  xdst1[0] = xcv ? (__bf16)xsrc1[0] : (__bf16)1.0f;
  __syncthreads();

  // ---- hoist loop-invariant WMMA B-fragments into VGPRs ----
  v16bf bh[4][4];   // W_hh  [gate][k-step]   (128 VGPRs)
  v16bf bx[4];      // W_ih+bias [gate]       ( 32 VGPRs)
#pragma unroll
  for (int g = 0; g < 4; ++g) {
#pragma unroll
    for (int ks = 0; ks < 4; ++ks)
      bh[g][ks] = load_b_frag(ldsWhh + (g * H + gcol) * H_PAD, ks * 32, hi);
    bx[g] = load_b_frag(ldsWih + (g * H + gcol) * INP_PAD, 0, hi);
  }

  // ---- recurrent loop: c in VGPRs, h double-buffered through LDS ----
  v8f cst;
#pragma unroll
  for (int r = 0; r < 8; ++r) cst[r] = 0.0f;

  const __bf16* aRowH0 = ldsH + (mt * 16 + mrow) * H_PAD;                  // buf 0
  const __bf16* aRowH1 = aRowH0 + MTILE * H_PAD;                           // buf 1
  const int hStoreBase0 = (mt * 16) * H_PAD + gcol;

  for (int t = 0; t < TSEQ; ++t) {
    const int cur = t & 1, nxt = cur ^ 1;
    // prefetch x[t+1] into the other x buffer (overlaps the WMMAs below)
    if (t + 1 < TSEQ) {
      const int toff = (t + 1) * IN;
      const int doff = nxt * MTILE * INP_PAD;
      xdst0[doff] = xcv ? (__bf16)xsrc0[toff] : (__bf16)1.0f;
      xdst1[doff] = xcv ? (__bf16)xsrc1[toff] : (__bf16)1.0f;
    }

    // x-projection + bias (K=32), accumulators start from inline-0 C
    v8f acc0 = {}, acc1 = {}, acc2 = {}, acc3 = {};
    {
      const __bf16* arow = ldsX + (cur * MTILE + mt * 16 + mrow) * INP_PAD;
      v16bf a = load_a_frag(arow, 0, hi);
      acc0 = WMMA_BF16(a, bx[0], acc0);
      acc1 = WMMA_BF16(a, bx[1], acc1);
      acc2 = WMMA_BF16(a, bx[2], acc2);
      acc3 = WMMA_BF16(a, bx[3], acc3);
    }

    // h(t-1) @ W_hh^T : read A from buffer (t+1)&1 = buffer holding h(t-1)
    const __bf16* aRowH = (nxt == 0) ? aRowH0 : aRowH1;
#pragma unroll
    for (int ks = 0; ks < 4; ++ks) {
      v16bf a = load_a_frag(aRowH, ks * 32, hi);
      acc0 = WMMA_BF16(a, bh[0][ks], acc0);
      acc1 = WMMA_BF16(a, bh[1][ks], acc1);
      acc2 = WMMA_BF16(a, bh[2][ks], acc2);
      acc3 = WMMA_BF16(a, bh[3][ks], acc3);
    }

    // gate nonlinearities + cell update (wave-local; hardware tanh)
    v8f hnew;
#pragma unroll
    for (int r = 0; r < 8; ++r) {
      float iv = fast_sigm(acc0[r]);
      float fv = fast_sigm(acc1[r]);
      float gv = fast_tanh(acc2[r]);
      float ov = fast_sigm(acc3[r]);
      float cc = fmaf(fv, cst[r], iv * gv);
      cst[r] = cc;
      hnew[r] = ov * fast_tanh(cc);
    }

    // write h(t) into buffer t&1 (nobody reads it this step)
    {
      __bf16* dst = ldsH + cur * MTILE * H_PAD + hStoreBase0;
#pragma unroll
      for (int r = 0; r < 8; ++r)
        dst[(r + 8 * hi) * H_PAD] = (__bf16)hnew[r];
    }
    __syncthreads();   // h(t) visible before step t+1 reads buffer t&1
  }

  // ---- head: y = relu(h @ W0^T + b0) @ W1^T + b1 ----
  // final h lives in buffer (TSEQ-1)&1 == 1
  const __bf16* aRowHF = aRowH1;

  __bf16* ldsW0 = ldsWih;  // reuse (128*H_PAD fits in 512*INP_PAD)
  for (int idx = tid; idx < H * H; idx += THREADS) {
    int r = idx >> 7, c = idx & 127;
    ldsW0[r * H_PAD + c] = (__bf16)W0[idx];
  }
  for (int idx = tid; idx < H; idx += THREADS) ldsBias[idx] = b0[idx];
  for (int idx = tid; idx < 16 * H; idx += THREADS) {
    int r = idx >> 7, c = idx & 127;
    ldsW1[r * H_PAD + c] = (r < NOUT) ? (__bf16)W1[r * H + c] : (__bf16)0.0f;
  }
  for (int idx = tid; idx < 16; idx += THREADS)
    ldsB1[idx] = (idx < NOUT) ? b1[idx] : 0.0f;
  __syncthreads();

  // layer 0: wave (mt, ws) computes rows mt*16.. , cols ws*16..
  {
    v8f acc;
    float bb = ldsBias[gcol];
#pragma unroll
    for (int r = 0; r < 8; ++r) acc[r] = bb;
#pragma unroll
    for (int k0 = 0; k0 < H; k0 += 32) {
      v16bf a = load_a_frag(aRowHF, k0, hi);
      v16bf b = load_b_frag(ldsW0 + gcol * H_PAD, k0, hi);
      acc = WMMA_BF16(a, b, acc);
    }
#pragma unroll
    for (int r = 0; r < 8; ++r)
      ldsY0[(mt * 16 + r + 8 * hi) * H_PAD + gcol] = (__bf16)fmaxf(acc[r], 0.0f);
  }
  __syncthreads();

  // layer 1: one wave per M-tile; N-tile 0 covers the 11 outputs (padded to 16)
  if (ws == 0) {
    v8f acc;
    float bb = ldsB1[ncol];
#pragma unroll
    for (int r = 0; r < 8; ++r) acc[r] = bb;
    const __bf16* aRowY = ldsY0 + (mt * 16 + mrow) * H_PAD;
#pragma unroll
    for (int k0 = 0; k0 < H; k0 += 32) {
      v16bf a = load_a_frag(aRowY, k0, hi);
      v16bf b = load_b_frag(ldsW1 + ncol * H_PAD, k0, hi);
      acc = WMMA_BF16(a, b, acc);
    }
    if (ncol < NOUT) {
#pragma unroll
      for (int r = 0; r < 8; ++r) {
        int row = blockIdx.x * MTILE + mt * 16 + r + 8 * hi;
        out[row * NOUT + ncol] = acc[r];
      }
    }
  }
}

extern "C" void kernel_launch(void* const* d_in, const int* in_sizes, int n_in,
                              void* d_out, int out_size, void* d_ws, size_t ws_size,
                              hipStream_t stream) {
  const float* x    = (const float*)d_in[0];
  const float* W_ih = (const float*)d_in[1];
  const float* W_hh = (const float*)d_in[2];
  const float* b_ih = (const float*)d_in[3];
  const float* b_hh = (const float*)d_in[4];
  const float* W0   = (const float*)d_in[5];
  const float* b0   = (const float*)d_in[6];
  const float* W1   = (const float*)d_in[7];
  const float* b1   = (const float*)d_in[8];

  dim3 grid(4096 / MTILE);   // 128 workgroups, one per 32 batch rows
  dim3 block(THREADS);       // 16 wave32 waves
  hipLaunchKernelGGL(lstm_fused_kernel, grid, block, SMEM_BYTES, stream,
                     x, W_ih, W_hh, b_ih, b_hh, W0, b0, W1, b1, (float*)d_out);
}
